// Fai_score_7653631721539
// MI455X (gfx1250) — compile-verified, matrix operand order checked by reading, not run on hardware
//
#include <hip/hip_runtime.h>

typedef __attribute__((ext_vector_type(16))) __bf16 v16bf;
typedef __attribute__((ext_vector_type(8)))  float  v8f;

#define F_NUM 256
#define E_DIM 300
#define E_PAD 320
#define K_TOT 640
#define K_STEPS 20
#define N_ENT 1024
#define M_MEN 256

// ---------------------------------------------------------------------------
// Prep: W [F,2,300] f32 -> Wb [F,640] bf16, zero-padded (300..319 per half)
// ---------------------------------------------------------------------------
__global__ void prep_w_kernel(const float* __restrict__ W, __bf16* __restrict__ Wb) {
  int f = blockIdx.x;
  for (int k = threadIdx.x; k < K_TOT; k += blockDim.x) {
    int part = (k >= E_PAD) ? 1 : 0;
    int e = k - part * E_PAD;
    float v = (e < E_DIM) ? W[(f * 2 + part) * E_DIM + e] : 0.0f;
    Wb[(size_t)f * K_TOT + k] = (__bf16)v;
  }
}

__global__ void zero_kernel(float* __restrict__ p, int n) {
  int i = blockIdx.x * blockDim.x + threadIdx.x;
  if (i < n) p[i] = 0.0f;
}

// ---------------------------------------------------------------------------
// Conv(kw=2 over full E) + bias + ReLU + mean-over-T, via bf16 WMMA.
// Grid: (ceil(T/16), B). Block: 256 threads = 8 waves; wave w owns filter
// tiles [32w..32w+15] and [32w+16..32w+31]. A tile rows = time steps.
// ---------------------------------------------------------------------------
__global__ void conv_pool_wmma_kernel(const float* __restrict__ X, int L,
                                      const __bf16* __restrict__ Wb,
                                      const float* __restrict__ bias,
                                      float* __restrict__ P) {
  const int T = L - 1;
  const int t0 = blockIdx.x * 16;
  const int b  = blockIdx.y;
  const float invT = 1.0f / (float)T;

  __shared__ __bf16 sX[17 * E_PAD];

  // Stage rows t0..t0+16 of X[b] into LDS as bf16, padded to 320 wide.
  const float* Xb = X + (size_t)b * L * E_DIM;
  for (int idx = threadIdx.x; idx < 17 * E_PAD; idx += blockDim.x) {
    int row = idx / E_PAD;
    int e   = idx - row * E_PAD;
    int t   = t0 + row;
    float v = (e < E_DIM && t < L) ? Xb[(size_t)t * E_DIM + e] : 0.0f;
    sX[idx] = (__bf16)v;
  }
  __syncthreads();

  const int lane = threadIdx.x & 31;
  const int wave = threadIdx.x >> 5;
  const int hi   = lane >> 4;      // 0 = lanes 0-15, 1 = lanes 16-31
  const int l16  = lane & 15;

  v8f acc0 = {}; v8f acc1 = {};
  const int f0 = wave * 32;
  const int f1 = f0 + 16;

  const __bf16* w0p = Wb + (size_t)(f0 + l16) * K_TOT;
  const __bf16* w1p = Wb + (size_t)(f1 + l16) * K_TOT;

  for (int ks = 0; ks < K_STEPS; ++ks) {
    const int ka = ks * 32 + hi * 8;   // K group 1 (elems 0..7)
    const int kc = ka + 16;            // K group 2 (elems 8..15)
    // A fragment: row M = l16 (time), conv row-concat mapping k -> (part, e)
    int pA = (ka >= E_PAD) ? 1 : 0; int eA = ka - pA * E_PAD;
    int pC = (kc >= E_PAD) ? 1 : 0; int eC = kc - pC * E_PAD;
    const __bf16* pa = &sX[(l16 + pA) * E_PAD + eA];
    const __bf16* pc = &sX[(l16 + pC) * E_PAD + eC];
    v16bf a;
#pragma unroll
    for (int i = 0; i < 8; ++i) { a[i] = pa[i]; a[i + 8] = pc[i]; }
    // B fragments: column N = filter f, pre-padded bf16 weight rows
    v16bf b0, b1;
#pragma unroll
    for (int i = 0; i < 8; ++i) {
      b0[i] = w0p[ka + i]; b0[i + 8] = w0p[kc + i];
      b1[i] = w1p[ka + i]; b1[i + 8] = w1p[kc + i];
    }
    acc0 = __builtin_amdgcn_wmma_f32_16x16x32_bf16(false, a, false, b0, (short)0, acc0, false, false);
    acc1 = __builtin_amdgcn_wmma_f32_16x16x32_bf16(false, a, false, b1, (short)0, acc1, false, false);
  }

  // Epilogue: bias + ReLU, branchless row mask (t >= T), mean into P[b,f].
  float bv0 = bias[f0 + l16];
  float bv1 = bias[f1 + l16];
  float s0 = 0.0f, s1 = 0.0f;
#pragma unroll
  for (int i = 0; i < 8; ++i) {
    int t = t0 + i + hi * 8;           // C layout: VGPR i -> M = i + 8*hi
    float valid = (t < T) ? 1.0f : 0.0f;
    float y0 = fmaxf(acc0[i] + bv0, 0.0f);
    float y1 = fmaxf(acc1[i] + bv1, 0.0f);
    s0 = fmaf(valid, y0, s0);
    s1 = fmaf(valid, y1, s1);
  }
  atomicAdd(&P[(size_t)b * F_NUM + f0 + l16], s0 * invT);
  atomicAdd(&P[(size_t)b * F_NUM + f1 + l16], s1 * invT);
}

// ---------------------------------------------------------------------------
// Row L2-normalize [rows,256] -> bf16 (and optional f32 copy)
// ---------------------------------------------------------------------------
__global__ void normalize_kernel(const float* __restrict__ P, __bf16* __restrict__ outb,
                                 float* __restrict__ outf) {
  __shared__ float red[256];
  int r = blockIdx.x, tid = threadIdx.x;
  float v = P[(size_t)r * F_NUM + tid];
  red[tid] = v * v;
  __syncthreads();
  for (int s = 128; s > 0; s >>= 1) {
    if (tid < s) red[tid] += red[tid + s];
    __syncthreads();
  }
  float nv = v * rsqrtf(red[0]);
  outb[(size_t)r * F_NUM + tid] = (__bf16)nv;
  if (outf) outf[(size_t)r * F_NUM + tid] = nv;
}

// ---------------------------------------------------------------------------
// dt[r] = dot(nx[r,:], nmd) for the broadcast cosine columns
// ---------------------------------------------------------------------------
__global__ void dot_md_kernel(const __bf16* __restrict__ nx, const float* __restrict__ nmd,
                              float* __restrict__ outv) {
  __shared__ float red[256];
  int r = blockIdx.x, tid = threadIdx.x;
  red[tid] = (float)nx[(size_t)r * F_NUM + tid] * nmd[tid];
  __syncthreads();
  for (int s = 128; s > 0; s >>= 1) {
    if (tid < s) red[tid] += red[tid + s];
    __syncthreads();
  }
  if (tid == 0) outv[r] = red[0];
}

// ---------------------------------------------------------------------------
// 4 cosine GEMMs [256x1024, K=256] fused with feats->score epilogue.
// Grid (8,16): blockIdx.x covers 128 n-cols (8 waves x 16), blockIdx.y 16 m-rows.
// ---------------------------------------------------------------------------
__global__ void score_wmma_kernel(const __bf16* __restrict__ nms, const __bf16* __restrict__ nmc,
                                  const __bf16* __restrict__ net, const __bf16* __restrict__ neb,
                                  const float* __restrict__ dt, const float* __restrict__ db,
                                  const float* __restrict__ w_local, const float* __restrict__ b_local,
                                  const int* __restrict__ mask, float* __restrict__ score) {
  const int lane = threadIdx.x & 31;
  const int wave = threadIdx.x >> 5;
  const int hi   = lane >> 4;
  const int l16  = lane & 15;

  const int nbase = blockIdx.x * 128 + wave * 16;
  const int mbase = blockIdx.y * 16;

  v8f a_st = {}, a_ct = {}, a_sb = {}, a_cb = {};

  const __bf16* pms = nms + (size_t)(mbase + l16) * F_NUM;
  const __bf16* pmc = nmc + (size_t)(mbase + l16) * F_NUM;
  const __bf16* pet = net + (size_t)(nbase + l16) * F_NUM;
  const __bf16* peb = neb + (size_t)(nbase + l16) * F_NUM;

  for (int ks = 0; ks < 8; ++ks) {
    const int ka = ks * 32 + hi * 8;
    const int kc = ka + 16;
    v16bf as, ac, bt, bb;
#pragma unroll
    for (int i = 0; i < 8; ++i) {
      as[i] = pms[ka + i]; as[i + 8] = pms[kc + i];
      ac[i] = pmc[ka + i]; ac[i + 8] = pmc[kc + i];
      bt[i] = pet[ka + i]; bt[i + 8] = pet[kc + i];
      bb[i] = peb[ka + i]; bb[i + 8] = peb[kc + i];
    }
    a_st = __builtin_amdgcn_wmma_f32_16x16x32_bf16(false, as, false, bt, (short)0, a_st, false, false);
    a_ct = __builtin_amdgcn_wmma_f32_16x16x32_bf16(false, ac, false, bt, (short)0, a_ct, false, false);
    a_sb = __builtin_amdgcn_wmma_f32_16x16x32_bf16(false, as, false, bb, (short)0, a_sb, false, false);
    a_cb = __builtin_amdgcn_wmma_f32_16x16x32_bf16(false, ac, false, bb, (short)0, a_cb, false, false);
  }

  const int n = nbase + l16;
  const float w0 = w_local[0], w1 = w_local[1], w2 = w_local[2];
  const float w3 = w_local[3], w4 = w_local[4], w5 = w_local[5];
  const float bl = b_local[0];
  const float gdt = 0.5f * dt[n] + 0.5f;
  const float gdb = 0.5f * db[n] + 0.5f;
#pragma unroll
  for (int i = 0; i < 8; ++i) {
    int m = mbase + i + hi * 8;
    int mk = mask[(size_t)m * N_ENT + n];
    float s = w0 * (0.5f * a_st[i] + 0.5f)
            + w1 * gdt
            + w2 * (0.5f * a_ct[i] + 0.5f)
            + w3 * (0.5f * a_sb[i] + 0.5f)
            + w4 * gdb
            + w5 * (0.5f * a_cb[i] + 0.5f)
            + bl;
    score[(size_t)m * N_ENT + n] = mk ? s : 0.0f;
  }
}

// ---------------------------------------------------------------------------
// Masked softmax with temperature 0.1 (matches reference: no max-subtract)
// ---------------------------------------------------------------------------
__global__ void softmax_kernel(const float* __restrict__ score, const int* __restrict__ mask,
                               float* __restrict__ normout) {
  __shared__ float red[256];
  int m = blockIdx.x, tid = threadIdx.x;
  float ev[4]; float part = 0.0f;
#pragma unroll
  for (int j = 0; j < 4; ++j) {
    int n = tid + j * 256;
    float s = score[(size_t)m * N_ENT + n];
    int mk = mask[(size_t)m * N_ENT + n];
    float e = mk ? __expf(s * 10.0f) : 0.0f;
    ev[j] = e; part += e;
  }
  red[tid] = part; __syncthreads();
  for (int st = 128; st > 0; st >>= 1) {
    if (tid < st) red[tid] += red[tid + st];
    __syncthreads();
  }
  float inv = 1.0f / red[0];
#pragma unroll
  for (int j = 0; j < 4; ++j) {
    int n = tid + j * 256;
    normout[(size_t)m * N_ENT + n] = ev[j] * inv;
  }
}

// ---------------------------------------------------------------------------
extern "C" void kernel_launch(void* const* d_in, const int* in_sizes, int n_in,
                              void* d_out, int out_size, void* d_ws, size_t ws_size,
                              hipStream_t stream) {
  const float* mention_vec = (const float*)d_in[0];   // [256,10,300]
  const float* context_vec = (const float*)d_in[1];   // [256,20,300]
  const float* doc_vec     = (const float*)d_in[2];   // [200,300]
  const float* title_vec   = (const float*)d_in[3];   // [1024,20,300]
  const float* body_vec    = (const float*)d_in[4];   // [1024,200,300]
  const float* W_ms = (const float*)d_in[5];  const float* b_ms = (const float*)d_in[6];
  const float* W_mc = (const float*)d_in[7];  const float* b_mc = (const float*)d_in[8];
  const float* W_md = (const float*)d_in[9];  const float* b_md = (const float*)d_in[10];
  const float* W_et = (const float*)d_in[11]; const float* b_et = (const float*)d_in[12];
  const float* W_eb = (const float*)d_in[13]; const float* b_eb = (const float*)d_in[14];
  const float* w_local = (const float*)d_in[15];
  const float* b_local = (const float*)d_in[16];
  const int*   mention_entity = (const int*)d_in[17]; // [256,1024]

  char* ws = (char*)d_ws;
  size_t off = 0;
  auto carve = [&](size_t bytes) -> void* {
    void* p = ws + off;
    off += (bytes + 255) & ~(size_t)255;
    return p;
  };

  __bf16* Wb_ms = (__bf16*)carve((size_t)F_NUM * K_TOT * 2);
  __bf16* Wb_mc = (__bf16*)carve((size_t)F_NUM * K_TOT * 2);
  __bf16* Wb_md = (__bf16*)carve((size_t)F_NUM * K_TOT * 2);
  __bf16* Wb_et = (__bf16*)carve((size_t)F_NUM * K_TOT * 2);
  __bf16* Wb_eb = (__bf16*)carve((size_t)F_NUM * K_TOT * 2);

  float* P_ms = (float*)carve((size_t)M_MEN * F_NUM * 4);
  float* P_mc = (float*)carve((size_t)M_MEN * F_NUM * 4);
  float* P_md = (float*)carve((size_t)F_NUM * 4);
  float* P_et = (float*)carve((size_t)N_ENT * F_NUM * 4);
  float* P_eb = (float*)carve((size_t)N_ENT * F_NUM * 4);

  __bf16* nms_b = (__bf16*)carve((size_t)M_MEN * F_NUM * 2);
  __bf16* nmc_b = (__bf16*)carve((size_t)M_MEN * F_NUM * 2);
  __bf16* nmd_b = (__bf16*)carve((size_t)F_NUM * 2);
  __bf16* net_b = (__bf16*)carve((size_t)N_ENT * F_NUM * 2);
  __bf16* neb_b = (__bf16*)carve((size_t)N_ENT * F_NUM * 2);
  float*  nmd_f = (float*)carve((size_t)F_NUM * 4);
  float*  dt_v  = (float*)carve((size_t)N_ENT * 4);
  float*  db_v  = (float*)carve((size_t)N_ENT * 4);

  float* score_out = (float*)d_out;                       // [256,1024]
  float* norm_out  = (float*)d_out + (size_t)M_MEN * N_ENT;

  // --- weight prep (fp32 -> padded bf16) ---
  prep_w_kernel<<<F_NUM, 256, 0, stream>>>(W_ms, Wb_ms);
  prep_w_kernel<<<F_NUM, 256, 0, stream>>>(W_mc, Wb_mc);
  prep_w_kernel<<<F_NUM, 256, 0, stream>>>(W_md, Wb_md);
  prep_w_kernel<<<F_NUM, 256, 0, stream>>>(W_et, Wb_et);
  prep_w_kernel<<<F_NUM, 256, 0, stream>>>(W_eb, Wb_eb);

  // --- zero pooled-feature accumulators (atomic targets) ---
  zero_kernel<<<(M_MEN * F_NUM + 255) / 256, 256, 0, stream>>>(P_ms, M_MEN * F_NUM);
  zero_kernel<<<(M_MEN * F_NUM + 255) / 256, 256, 0, stream>>>(P_mc, M_MEN * F_NUM);
  zero_kernel<<<(F_NUM + 255) / 256, 256, 0, stream>>>(P_md, F_NUM);
  zero_kernel<<<(N_ENT * F_NUM + 255) / 256, 256, 0, stream>>>(P_et, N_ENT * F_NUM);
  zero_kernel<<<(N_ENT * F_NUM + 255) / 256, 256, 0, stream>>>(P_eb, N_ENT * F_NUM);

  // --- conv + relu + mean pooling (WMMA GEMMs) ---
  conv_pool_wmma_kernel<<<dim3(13, 1024), 256, 0, stream>>>(body_vec, 200, Wb_eb, b_eb, P_eb);
  conv_pool_wmma_kernel<<<dim3(2, 1024), 256, 0, stream>>>(title_vec, 20, Wb_et, b_et, P_et);
  conv_pool_wmma_kernel<<<dim3(2, 256), 256, 0, stream>>>(context_vec, 20, Wb_mc, b_mc, P_mc);
  conv_pool_wmma_kernel<<<dim3(1, 256), 256, 0, stream>>>(mention_vec, 10, Wb_ms, b_ms, P_ms);
  conv_pool_wmma_kernel<<<dim3(13, 1), 256, 0, stream>>>(doc_vec, 200, Wb_md, b_md, P_md);

  // --- normalize rows ---
  normalize_kernel<<<M_MEN, 256, 0, stream>>>(P_ms, nms_b, nullptr);
  normalize_kernel<<<M_MEN, 256, 0, stream>>>(P_mc, nmc_b, nullptr);
  normalize_kernel<<<1, 256, 0, stream>>>(P_md, nmd_b, nmd_f);
  normalize_kernel<<<N_ENT, 256, 0, stream>>>(P_et, net_b, nullptr);
  normalize_kernel<<<N_ENT, 256, 0, stream>>>(P_eb, neb_b, nullptr);

  // --- broadcast cosines against md ---
  dot_md_kernel<<<N_ENT, 256, 0, stream>>>(net_b, nmd_f, dt_v);
  dot_md_kernel<<<N_ENT, 256, 0, stream>>>(neb_b, nmd_f, db_v);

  // --- 4 cosine GEMMs fused with feature linear + mask -> score ---
  score_wmma_kernel<<<dim3(8, 16), 256, 0, stream>>>(nms_b, nmc_b, net_b, neb_b,
                                                     dt_v, db_v, w_local, b_local,
                                                     mention_entity, score_out);

  // --- masked softmax (temp 0.1) -> norm ---
  softmax_kernel<<<M_MEN, 256, 0, stream>>>(score_out, mention_entity, norm_out);
}
